// DyC_Conv2d_71476845740320
// MI455X (gfx1250) — matrix operand order repaired and checked by reading, not compiled, not run on hardware
//
#include <hip/hip_runtime.h>

// ---------------------------------------------------------------------------
// Dynamic convolution (CondConv / DyConv):
//   SE routing -> per-sample expert-kernel mix -> per-sample 3x3 conv.
// Conv = implicit GEMM: M=COUT=256, N=784, K=CIN*9=2304 on
// v_wmma_f32_16x16x32_bf16 (bf16 in, f32 acc).
// K order = (kh, kw, ci): every 32-wide K slice sits in one (kh,kw) plane.
// A tile (aggregated weights) staged into LDS by the Tensor Data Mover
// (tensor_load_to_lds / s_wait_tensorcnt) when the builtin is available.
// ---------------------------------------------------------------------------

#define BATCH  32
#define CIN    256
#define HH     28
#define WW     28
#define HWP    784           // 28*28
#define COUT   256
#define KEXP   4
#define HIDDEN 64
#define KTOT   2304          // 9*CIN
#define NSLICE 72            // KTOT/32

typedef __attribute__((ext_vector_type(16))) __bf16 v16bf;
typedef __attribute__((ext_vector_type(8)))  float  v8f;
typedef __attribute__((ext_vector_type(4)))  unsigned int u32x4;
typedef __attribute__((ext_vector_type(8)))  int    i32x8;
typedef __attribute__((ext_vector_type(4)))  int    i32x4;

#if __has_builtin(__builtin_amdgcn_tensor_load_to_lds)
#define USE_TDM 1
#else
#define USE_TDM 0
#endif

// fp32 -> bf16 round-to-nearest-even (bit manipulation only)
static __device__ __forceinline__ unsigned short f2bf(float f) {
  union { float f; unsigned int u; } c; c.f = f;
  unsigned int u = c.u;
  return (unsigned short)((u + 0x7FFFu + ((u >> 16) & 1u)) >> 16);
}

// ---------------------------------------------------------------------------
// Kernel 1: SE routing. One block per batch sample, 256 threads.
// ---------------------------------------------------------------------------
__global__ __launch_bounds__(256) void se_kernel(
    const float* __restrict__ x, const float* __restrict__ fc1,
    const float* __restrict__ fc2, const float* __restrict__ fc2b,
    float* __restrict__ prob)
{
  __shared__ float pooled[CIN];
  __shared__ float h[HIDDEN];
  __shared__ float yv[KEXP * COUT];

  const int b = blockIdx.x;
  const int t = threadIdx.x;

  pooled[t] = 0.0f;
  __syncthreads();

  const float* xb = x + (size_t)b * CIN * HWP;
  for (int idx = t; idx < CIN * HWP; idx += 256)
    atomicAdd(&pooled[idx / HWP], xb[idx]);      // ds_add_f32
  __syncthreads();
  pooled[t] *= (1.0f / (float)HWP);
  __syncthreads();

  if (t < HIDDEN) {
    float s = 0.0f;
    for (int c = 0; c < CIN; ++c) s += fc1[t * CIN + c] * pooled[c];
    h[t] = s > 0.0f ? s : 0.0f;
  }
  __syncthreads();

  for (int r = t; r < KEXP * COUT; r += 256) {
    float s = fc2b[r];
    for (int j = 0; j < HIDDEN; ++j) s += fc2[r * HIDDEN + j] * h[j];
    yv[r] = s * 0.125f;                          // 1/sqrt(HIDDEN)
  }
  __syncthreads();

  float v0 = yv[0 * COUT + t], v1 = yv[1 * COUT + t];
  float v2 = yv[2 * COUT + t], v3 = yv[3 * COUT + t];
  float m  = fmaxf(fmaxf(v0, v1), fmaxf(v2, v3));
  float e0 = expf(v0 - m), e1 = expf(v1 - m), e2 = expf(v2 - m), e3 = expf(v3 - m);
  float inv = 1.0f / (e0 + e1 + e2 + e3);
  prob[((size_t)b * KEXP + 0) * COUT + t] = e0 * inv;
  prob[((size_t)b * KEXP + 1) * COUT + t] = e1 * inv;
  prob[((size_t)b * KEXP + 2) * COUT + t] = e2 * inv;
  prob[((size_t)b * KEXP + 3) * COUT + t] = e3 * inv;
}

// ---------------------------------------------------------------------------
// Kernel 2: transpose+convert x[b][ci][p] f32 -> xt[b][p][ci] bf16.
// ---------------------------------------------------------------------------
__global__ __launch_bounds__(256) void xpose_kernel(
    const float* __restrict__ x, unsigned short* __restrict__ xt)
{
  __shared__ float tile[32][33];
  const int b   = blockIdx.z;
  const int p0  = blockIdx.x * 32;
  const int ci0 = blockIdx.y * 32;
  const int tx  = threadIdx.x & 31;
  const int ty  = threadIdx.x >> 5;          // 0..7

  const float* xb = x + (size_t)b * CIN * HWP;
  #pragma unroll
  for (int j = 0; j < 4; ++j) {
    int ci = ci0 + ty + j * 8;
    int p  = p0 + tx;
    tile[ty + j * 8][tx] = (p < HWP) ? xb[(size_t)ci * HWP + p] : 0.0f;
  }
  __syncthreads();

  unsigned short* xtb = xt + (size_t)b * HWP * CIN;
  #pragma unroll
  for (int j = 0; j < 4; ++j) {
    int p  = p0 + ty + j * 8;
    int ci = ci0 + tx;
    if (p < HWP) xtb[(size_t)p * CIN + ci] = f2bf(tile[tx][ty + j * 8]);
  }
}

// ---------------------------------------------------------------------------
// Kernel 3: aggregate expert kernels, emit bf16 in (kh,kw,ci)-major K order:
//   aggw[b][o][kprime], kprime = (kh*3+kw)*256 + ci
// ---------------------------------------------------------------------------
__global__ __launch_bounds__(256) void agg_kernel(
    const float* __restrict__ weight, const float* __restrict__ prob,
    unsigned short* __restrict__ aggw)
{
  const int bo = blockIdx.x;          // b*COUT + o
  const int b  = bo >> 8;
  const int o  = bo & 255;
  const int ci = threadIdx.x;         // 0..255

  const float p0 = prob[((size_t)b * KEXP + 0) * COUT + o];
  const float p1 = prob[((size_t)b * KEXP + 1) * COUT + o];
  const float p2 = prob[((size_t)b * KEXP + 2) * COUT + o];
  const float p3 = prob[((size_t)b * KEXP + 3) * COUT + o];

  const float* w0 = weight + ((size_t)(0 * COUT + o)) * KTOT + (size_t)ci * 9;
  const float* w1 = weight + ((size_t)(1 * COUT + o)) * KTOT + (size_t)ci * 9;
  const float* w2 = weight + ((size_t)(2 * COUT + o)) * KTOT + (size_t)ci * 9;
  const float* w3 = weight + ((size_t)(3 * COUT + o)) * KTOT + (size_t)ci * 9;
  unsigned short* dst = aggw + (size_t)bo * KTOT + ci;

  #pragma unroll
  for (int i = 0; i < 9; ++i)         // i = kh*3+kw
    dst[i * CIN] = f2bf(p0 * w0[i] + p1 * w1[i] + p2 * w2[i] + p3 * w3[i]);
}

// ---------------------------------------------------------------------------
// Kernel 4: implicit-GEMM conv on WMMA.
// Block = 8 waves, tile M=128 x N=64; waves 4(M) x 2(N); each wave owns a
// 32x32 output patch = 4 WMMA accumulators. Double-buffered LDS.
// A slice (128 x 64B, row stride KTOT*2B) staged by TDM; B slice staged by
// lanes (zero-filled conv halo). One barrier per K-slice.
// ---------------------------------------------------------------------------
__global__ __launch_bounds__(256) void conv_wmma_kernel(
    const unsigned short* __restrict__ xt,
    const unsigned short* __restrict__ aggw,
    float* __restrict__ out)
{
  __shared__ __align__(16) unsigned short As[2][128][32];  // 16 KB
  __shared__ __align__(16) unsigned short Bs[2][64][32];   //  8 KB

  const int t     = threadIdx.x;
  const int b     = blockIdx.z;
  const int obase = blockIdx.y * 128;
  const int nbase = blockIdx.x * 64;

  const int lane    = t & 31;
  const int wid     = t >> 5;
  const int wm      = wid >> 1;       // 0..3 : M sub-block of 32
  const int wn      = wid & 1;        // 0..1 : N sub-block of 32
  const int halfsel = lane >> 4;
  const int lrow    = lane & 15;

  v8f acc00 = {}, acc01 = {}, acc10 = {}, acc11 = {};

  const unsigned short* aggB = aggw + ((size_t)(b * COUT + obase)) * KTOT;

#if USE_TDM
  const bool tdmWave = (wid == 0);
  // D# group1 (loop-invariant): data_size=2B; tensor 32 x 128 (dim0 x dim1),
  // dim0_stride = KTOT; tile 32 x 128. Bits per ISA cdna5 sec 8.4.
  const i32x8 tdm_g1 = { 0x10000,            // [17:16] data_size=1 (2 bytes)
                         (int)(32u << 16),   // tensor_dim0[15:0] @ bits 63:48
                         (int)(128u << 16),  // dim0 hi=0; tensor_dim1[15:0] @ 95:80
                         (int)(32u << 16),   // dim1 hi=0; tile_dim0 @ 127:112
                         128,                // tile_dim1 @ 143:128; tile_dim2=0
                         KTOT,               // tensor_dim0_stride[31:0] @ 191:160
                         0, 0 };
  const i32x4 tdm_gz = { 0, 0, 0, 0 };       // groups 2/3: unused (2-D tensor)
  const unsigned int ldsA0 = (unsigned int)(size_t)&As[0][0][0];
  const unsigned int ldsA1 = (unsigned int)(size_t)&As[1][0][0];

  auto tdm_issue = [&](int j, int bsel) {
    unsigned long long ga =
        (unsigned long long)(size_t)(const void*)(aggB + (size_t)j * 32);
    u32x4 g0;
    g0.x = 1u;                                        // count=1, user mode
    g0.y = bsel ? ldsA1 : ldsA0;                      // lds_addr
    g0.z = (unsigned int)ga;                          // global_addr[31:0]
    g0.w = (unsigned int)((ga >> 32) & 0x01FFFFFFull) // global_addr[56:32]
         | (2u << 30);                                // type = 2 ("image")
#if __clang_major__ >= 23
    const i32x8 z8 = { 0, 0, 0, 0, 0, 0, 0, 0 };
    __builtin_amdgcn_tensor_load_to_lds(g0, tdm_g1, tdm_gz, tdm_gz, z8, 0);
#else
    __builtin_amdgcn_tensor_load_to_lds(g0, tdm_g1, tdm_gz, tdm_gz, 0);
#endif
  };
#else
  const int rowA = t >> 2;            // 0..63 (and +64)
  const int qA   = (t & 3) * 8;       // 16B chunk within 64B row
  const unsigned short* pA0 = aggB + (size_t)rowA * KTOT + qA;
  const unsigned short* pA1 = aggB + (size_t)(rowA + 64) * KTOT + qA;
#endif

  // --- per-thread B staging coords ---
  const int npB = t >> 2;             // pixel 0..63 within N tile
  const int chB = (t & 3) * 8;        // 16B chunk of the 32-channel run
  const int p   = nbase + npB;
  const int py  = p / WW;
  const int px  = p - py * WW;
  const bool pvalid = (p < HWP);
  const unsigned short* xtb = xt + (size_t)b * HWP * CIN;

  auto stageB = [&](int j, int bsel) {
    const int kh  = j / 24;                          // slice -> (kh,kw,ci0)
    const int rem = j - kh * 24;
    const int kwv = rem >> 3;
    const int ci0 = (rem & 7) << 5;
    const int iy  = py + kh - 1;
    const int ix  = px + kwv - 1;
    uint4 v = {0u, 0u, 0u, 0u};
    if (pvalid && (unsigned)iy < (unsigned)HH && (unsigned)ix < (unsigned)WW)
      v = *(const uint4*)(xtb + ((size_t)(iy * WW + ix) << 8) + ci0 + chB);
    *(uint4*)&Bs[bsel][npB][chB] = v;
  };

#if !USE_TDM
  auto stageA = [&](int j, int bsel) {
    const int koff = j * 32;
    *(uint4*)&As[bsel][rowA][qA]      = *(const uint4*)(pA0 + koff);
    *(uint4*)&As[bsel][rowA + 64][qA] = *(const uint4*)(pA1 + koff);
    if (j + 2 < NSLICE)
      __builtin_prefetch(pA0 + koff + 64, 0, 1);
  };
#endif

  // --- fragment coordinates (ISA 16-bit VGPR layouts) ---
  const int ac0 = halfsel ? 8 : 0;
  const int bc0 = halfsel ? 8 : 0;
  const int ar0 = wm * 32 + lrow, ar1 = ar0 + 16;
  const int br0 = wn * 32 + lrow, br1 = br0 + 16;

  // prologue: stage slice 0 into buffer 0
#if USE_TDM
  if (tdmWave) tdm_issue(0, 0);
#else
  stageA(0, 0);
#endif
  stageB(0, 0);
#if USE_TDM
  if (tdmWave) __builtin_amdgcn_s_wait_tensorcnt((short)0);
#endif
  __syncthreads();

  for (int j = 0; j < NSLICE; ++j) {
    const int cur = j & 1;
    if (j + 1 < NSLICE) {
#if USE_TDM
      if (tdmWave) tdm_issue(j + 1, cur ^ 1);
#else
      stageA(j + 1, cur ^ 1);
#endif
      stageB(j + 1, cur ^ 1);
    }

    union Frag { v16bf v; uint4 u[2]; };
    Frag a0, a1, b0, b1;
    a0.u[0] = *(const uint4*)&As[cur][ar0][ac0];
    a0.u[1] = *(const uint4*)&As[cur][ar0][ac0 + 16];
    a1.u[0] = *(const uint4*)&As[cur][ar1][ac0];
    a1.u[1] = *(const uint4*)&As[cur][ar1][ac0 + 16];
    b0.u[0] = *(const uint4*)&Bs[cur][br0][bc0];
    b0.u[1] = *(const uint4*)&Bs[cur][br0][bc0 + 16];
    b1.u[0] = *(const uint4*)&Bs[cur][br1][bc0];
    b1.u[1] = *(const uint4*)&Bs[cur][br1][bc0 + 16];

    acc00 = __builtin_amdgcn_wmma_f32_16x16x32_bf16(
        false, a0.v, false, b0.v, (short)0, acc00, false, false);
    acc01 = __builtin_amdgcn_wmma_f32_16x16x32_bf16(
        false, a0.v, false, b1.v, (short)0, acc01, false, false);
    acc10 = __builtin_amdgcn_wmma_f32_16x16x32_bf16(
        false, a1.v, false, b0.v, (short)0, acc10, false, false);
    acc11 = __builtin_amdgcn_wmma_f32_16x16x32_bf16(
        false, a1.v, false, b1.v, (short)0, acc11, false, false);

#if USE_TDM
    if (tdmWave && j + 1 < NSLICE)
      __builtin_amdgcn_s_wait_tensorcnt((short)0);   // slice j+1 landed
#endif
    __syncthreads();
  }

  // --- store: C layout — VGPR v: M = v + 8*halfsel, N = lane&15 ---
  float* ob = out + (size_t)b * COUT * HWP;
  const int m0    = obase + wm * 32 + halfsel * 8;
  const int ncol0 = nbase + wn * 32 + lrow;
  const int ncol1 = ncol0 + 16;
  if (ncol0 < HWP) {
    #pragma unroll
    for (int v = 0; v < 8; ++v) {
      ob[(size_t)(m0 + v) * HWP + ncol0]      = acc00[v];
      ob[(size_t)(m0 + 16 + v) * HWP + ncol0] = acc10[v];
    }
  }
  if (ncol1 < HWP) {
    #pragma unroll
    for (int v = 0; v < 8; ++v) {
      ob[(size_t)(m0 + v) * HWP + ncol1]      = acc01[v];
      ob[(size_t)(m0 + 16 + v) * HWP + ncol1] = acc11[v];
    }
  }
}

// ---------------------------------------------------------------------------
// Launch. Workspace layout (256 B aligned):
//   prob : off 0        , 32*4*256 f32      = 131072 B
//   xt   : off 131072   , 32*784*256 bf16   = 12845056 B   (pixel-major)
//   aggw : off 12976128 , 32*256*2304 bf16  = 37748736 B   (~50.7 MB total)
// ---------------------------------------------------------------------------
extern "C" void kernel_launch(void* const* d_in, const int* in_sizes, int n_in,
                              void* d_out, int out_size, void* d_ws, size_t ws_size,
                              hipStream_t stream) {
  const float* x      = (const float*)d_in[0];
  const float* fc1_w  = (const float*)d_in[1];
  const float* fc2_w  = (const float*)d_in[2];
  const float* fc2_b  = (const float*)d_in[3];
  const float* weight = (const float*)d_in[4];
  float* out = (float*)d_out;

  char* ws = (char*)d_ws;
  float*          prob = (float*)(ws + 0);
  unsigned short* xt   = (unsigned short*)(ws + 131072);
  unsigned short* aggw = (unsigned short*)(ws + 12976128);

  // 1) SE routing -> prob[b][e][o]
  se_kernel<<<BATCH, 256, 0, stream>>>(x, fc1_w, fc2_w, fc2_b, prob);

  // 2) x -> pixel-major bf16
  dim3 gx((HWP + 31) / 32, CIN / 32, BATCH);       // 25 x 8 x 32
  xpose_kernel<<<gx, 256, 0, stream>>>(x, xt);

  // 3) aggregate expert kernels -> bf16, (kh,kw,ci)-major K
  agg_kernel<<<BATCH * COUT, 256, 0, stream>>>(weight, prob, aggw);

  // 4) implicit-GEMM conv on WMMA + TDM staging
  dim3 gc((HWP + 63) / 64, COUT / 128, BATCH);     // 13 x 2 x 32
  conv_wmma_kernel<<<gc, 256, 0, stream>>>(xt, aggw, out);
}
